// LSTMChar_8418135900750
// MI455X (gfx1250) — compile-verified, matrix operand order checked by reading, not run on hardware
//
#include <hip/hip_runtime.h>
#include <cmath>

// ---------------------------------------------------------------------------
// Problem constants (match reference)
// ---------------------------------------------------------------------------
#define NW    8192     // words
#define TCH   16       // chars per word (padded)
#define CE    64       // char embedding dim
#define HE    128      // char-LSTM hidden
#define WE    256      // word embedding dim
#define HP    256      // word-LSTM hidden
#define TAGS  50
#define KE    (CE + HE)    // 192 : combined K for char LSTM
#define KP    (WE + HE)    // 384 : combined K for word input projection

// ---------------------------------------------------------------------------
// CDNA5 WMMA types & helpers (wave32, 16x16x32 bf16 -> f32)
// ---------------------------------------------------------------------------
typedef __attribute__((ext_vector_type(16))) __bf16        v16bf;
typedef __attribute__((ext_vector_type(8)))  float         v8f;
typedef __attribute__((ext_vector_type(4)))  unsigned int  v4u;
typedef __attribute__((ext_vector_type(4)))  float         v4f;
typedef __attribute__((ext_vector_type(2)))  float         v2f;

union Frag {            // POD union: fill via u[] / q[], feed .v to WMMA
    v16bf          v;
    unsigned short u[16];
    v4u            q[2];
};

__device__ __forceinline__ v8f wmma_bf16(v16bf a, v16bf b, v8f c) {
    return __builtin_amdgcn_wmma_f32_16x16x32_bf16(
        /*neg_a=*/false, a, /*neg_b=*/false, b,
        /*c_mod=*/(short)0, c, /*reuse_a=*/false, /*reuse_b=*/false);
}

// CDNA5 16-bit A/B fragment K index for element e in half 'hi' (lane>>4)
__device__ __forceinline__ int kmap(int e, int hi) {
    return e + ((e >> 3) << 3) + (hi << 3);   // e<8 ? e : e+8, then +hi*8
}

__device__ __forceinline__ unsigned short f2bf(float f) {   // RNE f32->bf16
    unsigned int x = __float_as_uint(f);
    unsigned int r = x + 0x7fffu + ((x >> 16) & 1u);
    return (unsigned short)(r >> 16);
}
__device__ __forceinline__ float bf2f(unsigned short u) {
    return __uint_as_float(((unsigned int)u) << 16);
}

// Branch-free fast activations: v_exp_f32 + v_rcp_f32, clamped (no Inf/NaN path)
__device__ __forceinline__ float sigm_fast(float x) {
    float xc = fminf(fmaxf(x, -30.f), 30.f);
    return __builtin_amdgcn_rcpf(1.f + __expf(-xc));
}
__device__ __forceinline__ float tanh_fast(float x) {
    float xc = fminf(fmaxf(x, -15.f), 15.f);
    float e  = __expf(2.f * xc);
    return (e - 1.f) * __builtin_amdgcn_rcpf(e + 1.f);
}

// packed f32 FMA of a bf16 pair (in one u32) against two h values
__device__ __forceinline__ v2f pkfma(unsigned int uw, float ha, float hb, v2f acc) {
    v2f w, h;
    w.x = __uint_as_float(uw << 16);
    w.y = __uint_as_float(uw & 0xffff0000u);
    h.x = ha; h.y = hb;
    return __builtin_elementwise_fma(w, h, acc);   // -> v_pk_fma_f32
}

// ---------------------------------------------------------------------------
// Workspace layout (bytes)
// ---------------------------------------------------------------------------
static constexpr size_t OFF_HCHAR = 0;                                     // f32 [NW][HE]
static constexpr size_t OFF_WFE   = OFF_HCHAR + (size_t)NW * HE * 4;       // bf16 frag 32*6*512
static constexpr size_t OFF_BE    = OFF_WFE   + (size_t)32 * 6 * 512 * 2;  // f32 [512]
static constexpr size_t OFF_FEATS = OFF_BE    + 512 * 4;                   // bf16 [NW][KP]
static constexpr size_t OFF_WFP   = OFF_FEATS + (size_t)NW * KP * 2;       // bf16 frag 64*12*512
static constexpr size_t OFF_BP    = OFF_WFP   + (size_t)64 * 12 * 512 * 2; // f32 [1024]
static constexpr size_t OFF_WHH   = OFF_BP    + 1024 * 4;                  // bf16 [1024][256]
static constexpr size_t OFF_XPROJ = OFF_WHH   + (size_t)1024 * 256 * 2;    // f32 [NW][1024]
static constexpr size_t OFF_HS    = OFF_XPROJ + (size_t)NW * 1024 * 4;     // f32 [NW][HP]

// ---------------------------------------------------------------------------
// Weight pre-shuffle: W (row-major [n][k]) -> WMMA B-fragment order:
//   flat = ((ntile*KC + kc)*32 + lane)*16 + e ;  value = W[ntile*16+(lane&15)]
//                                                         [kc*32 + kmap(e,lane>>4)]
// ---------------------------------------------------------------------------
__global__ void prep_wfrag_e(const float* __restrict__ Wih,
                             const float* __restrict__ Whh,
                             unsigned short* __restrict__ out) {
    int idx  = blockIdx.x * 256 + threadIdx.x;           // < 32*6*512 = 98304
    int e    = idx & 15;
    int lane = (idx >> 4) & 31;
    int kc   = (idx >> 9) % 6;
    int nt   = idx / 3072;
    int k    = kc * 32 + kmap(e, lane >> 4);
    int n    = nt * 16 + (lane & 15);
    float v  = (k < CE) ? Wih[n * CE + k] : Whh[n * HE + (k - CE)];
    out[idx] = f2bf(v);
}

__global__ void prep_wfrag_p(const float* __restrict__ Wih,
                             unsigned short* __restrict__ out) {
    int idx  = blockIdx.x * 256 + threadIdx.x;           // < 64*12*512 = 393216
    int e    = idx & 15;
    int lane = (idx >> 4) & 31;
    int kc   = (idx >> 9) % 12;
    int nt   = idx / 6144;
    int k    = kc * 32 + kmap(e, lane >> 4);
    int n    = nt * 16 + (lane & 15);
    out[idx] = f2bf(Wih[n * KP + k]);
}

__global__ void prep_misc(const float* __restrict__ Whh_p,
                          const float* __restrict__ bih_e, const float* __restrict__ bhh_e,
                          const float* __restrict__ bih_p, const float* __restrict__ bhh_p,
                          unsigned short* __restrict__ whh_bf,
                          float* __restrict__ bias_e, float* __restrict__ bias_p) {
    int idx = blockIdx.x * 256 + threadIdx.x;            // < 1024*256 = 262144
    whh_bf[idx] = f2bf(Whh_p[idx]);
    if (idx < 4 * HP)  bias_p[idx] = bih_p[idx] + bhh_p[idx];
    if (idx < 4 * HE)  bias_e[idx] = bih_e[idx] + bhh_e[idx];
}

// ---------------------------------------------------------------------------
// Phase A: batched char LSTM.  One block = 16 words, 8 waves.
// Wave w owns hidden slice [16w,16w+16): output tiles {w, w+8, w+16, w+24}
// are exactly the i/f/g/o gates for that slice, so cell update is register-local.
// B-fragments are loop-invariant -> compiler keeps them register-resident.
// ---------------------------------------------------------------------------
__global__ __launch_bounds__(256) void char_lstm_kernel(
    const float* __restrict__ char_emb, const int* __restrict__ char_ids,
    const int* __restrict__ char_len, const unsigned short* __restrict__ wfrag,
    const float* __restrict__ bias, float* __restrict__ h_char) {

    __shared__ unsigned short xh[16][200];   // [word][K=192 (+pad)]  x:0..63 h:64..191
    const int tid  = threadIdx.x;
    const int wid  = tid >> 5;
    const int lane = tid & 31;
    const int lo   = lane & 15;
    const int hi   = lane >> 4;
    const int wordBase = blockIdx.x * 16;

    for (int i = tid; i < 16 * 200; i += 256) (&xh[0][0])[i] = 0;

    float c[8], h[8];
    int   len[8];
#pragma unroll
    for (int v = 0; v < 8; ++v) {
        c[v] = 0.f; h[v] = 0.f;
        len[v] = char_len[wordBase + v + hi * 8];
    }
    const float b_i = bias[  0 + wid * 16 + lo];
    const float b_f = bias[128 + wid * 16 + lo];
    const float b_g = bias[256 + wid * 16 + lo];
    const float b_o = bias[384 + wid * 16 + lo];
    __syncthreads();

    for (int t = 0; t < TCH; ++t) {
        // gather this step's char embeddings -> bf16 LDS (16 words x 64)
        {
            int word = tid >> 4;
            int e4   = (tid & 15) * 4;
            int cid  = char_ids[(wordBase + word) * TCH + t];
            v4f cv = *reinterpret_cast<const v4f*>(char_emb + (size_t)cid * CE + e4);
            xh[word][e4 + 0] = f2bf(cv.x);
            xh[word][e4 + 1] = f2bf(cv.y);
            xh[word][e4 + 2] = f2bf(cv.z);
            xh[word][e4 + 3] = f2bf(cv.w);
        }
        __syncthreads();

        // A fragments: per-lane K-map is two contiguous runs of 8 -> 2x ds_load_b128
        Frag fa[6];
#pragma unroll
        for (int kc = 0; kc < 6; ++kc) {
            int k0 = kc * 32 + hi * 8;
            fa[kc].q[0] = *reinterpret_cast<const v4u*>(&xh[lo][k0]);
            fa[kc].q[1] = *reinterpret_cast<const v4u*>(&xh[lo][k0 + 16]);
        }
        __syncthreads();   // all reads done before h-region rewrite below

        v8f g[4];
#pragma unroll
        for (int gi = 0; gi < 4; ++gi) {
            v8f acc = {0.f, 0.f, 0.f, 0.f, 0.f, 0.f, 0.f, 0.f};
            int ntile = wid + 8 * gi;
#pragma unroll
            for (int kc = 0; kc < 6; ++kc) {
                Frag fb;
                const v4u* p = reinterpret_cast<const v4u*>(wfrag) +
                               ((size_t)(ntile * 6 + kc) * 32 + lane) * 2;
                fb.q[0] = p[0];
                fb.q[1] = p[1];
                acc = wmma_bf16(fa[kc].v, fb.v, acc);
            }
            g[gi] = acc;
        }

        // activations + masked state update; D layout: (word=v+hi*8, hid=16*wid+lo)
#pragma unroll
        for (int v = 0; v < 8; ++v) {
            float iv = sigm_fast(g[0][v] + b_i);
            float fv = sigm_fast(g[1][v] + b_f);
            float gv = tanh_fast(g[2][v] + b_g);
            float ov = sigm_fast(g[3][v] + b_o);
            float cn = fv * c[v] + iv * gv;
            float hn = ov * tanh_fast(cn);
            bool  m  = (t < len[v]);
            c[v] = m ? cn : c[v];
            h[v] = m ? hn : h[v];
            xh[v + hi * 8][CE + wid * 16 + lo] = f2bf(h[v]);
        }
        __syncthreads();
    }

#pragma unroll
    for (int v = 0; v < 8; ++v)
        h_char[(size_t)(wordBase + v + hi * 8) * HE + wid * 16 + lo] = h[v];
}

// ---------------------------------------------------------------------------
// feats[w] = concat(word_emb[sentence[w]] (256), h_char[w] (128)) -> bf16
// ---------------------------------------------------------------------------
__global__ __launch_bounds__(256) void feats_kernel(
    const float* __restrict__ word_emb, const float* __restrict__ h_char,
    const int* __restrict__ sentence, unsigned short* __restrict__ feats) {
    int idx = (blockIdx.x * 256 + threadIdx.x) * 4;      // < NW*KP
    int w = idx / KP, k = idx % KP;
    const float* src = (k < WE) ? (word_emb + (size_t)sentence[w] * WE + k)
                                : (h_char + (size_t)w * HE + (k - WE));
    v4f v = *reinterpret_cast<const v4f*>(src);
    unsigned short* d = feats + (size_t)w * KP + k;
    d[0] = f2bf(v.x); d[1] = f2bf(v.y); d[2] = f2bf(v.z); d[3] = f2bf(v.w);
}

// ---------------------------------------------------------------------------
// Phase B: xproj = feats @ Wih_p^T + (bih_p+bhh_p)   [8192,384]x[384,1024]
// Block = 16x128 output slab, wave per 16x16 tile, K=384 = 12 WMMAs.
// ---------------------------------------------------------------------------
__global__ __launch_bounds__(256) void xproj_kernel(
    const unsigned short* __restrict__ feats, const unsigned short* __restrict__ wfrag,
    const float* __restrict__ bias, float* __restrict__ xproj) {

    __shared__ unsigned short at[16][392];               // padded, conflict-free
    const int tid  = threadIdx.x;
    const int wid  = tid >> 5;
    const int lane = tid & 31;
    const int lo   = lane & 15;
    const int hi   = lane >> 4;
    const int mt    = blockIdx.x >> 3;
    const int ntile = (blockIdx.x & 7) * 8 + wid;

    for (int i = tid; i < 16 * 48; i += 256) {           // stage 16x384 bf16 tile
        int r = i / 48, cs = i % 48;
        *reinterpret_cast<v4u*>(&at[r][cs * 8]) =
            *reinterpret_cast<const v4u*>(feats + ((size_t)(mt * 16 + r)) * KP + cs * 8);
    }
    __syncthreads();

    v8f acc = {0.f, 0.f, 0.f, 0.f, 0.f, 0.f, 0.f, 0.f};
#pragma unroll
    for (int kc = 0; kc < 12; ++kc) {
        Frag fa, fb;
        int k0 = kc * 32 + hi * 8;
        fa.q[0] = *reinterpret_cast<const v4u*>(&at[lo][k0]);
        fa.q[1] = *reinterpret_cast<const v4u*>(&at[lo][k0 + 16]);
        const v4u* p = reinterpret_cast<const v4u*>(wfrag) +
                       ((size_t)(ntile * 12 + kc) * 32 + lane) * 2;
        fb.q[0] = p[0];
        fb.q[1] = p[1];
        acc = wmma_bf16(fa.v, fb.v, acc);
    }
    float b = bias[ntile * 16 + lo];
#pragma unroll
    for (int v = 0; v < 8; ++v)
        xproj[(size_t)(mt * 16 + v + hi * 8) * 1024 + ntile * 16 + lo] = acc[v] + b;
}

// ---------------------------------------------------------------------------
// Phase C: strictly sequential word LSTM.  Single persistent 1024-thread block.
// Thread j owns gate j.  Gates 0..511 read their Whh rows from LDS (256 KB of
// the CDNA5 320 KB WGP LDS, staged once); gates 512..1023 stream bf16 rows
// from L2.  Dot product runs as packed f32 FMAs (v_pk_fma_f32), h read 4-wide.
// ---------------------------------------------------------------------------
#define WLSTM_LDS_ROWS 512
static constexpr unsigned WLSTM_SMEM =
    (unsigned)WLSTM_LDS_ROWS * HP * 2 + (HP + HP + 4 * HP) * 4;   // 268288 B

__global__ __launch_bounds__(1024) void word_lstm_kernel(
    const float* __restrict__ xproj, const unsigned short* __restrict__ whh,
    float* __restrict__ hs) {

    extern __shared__ char smem[];
    unsigned short* wlds = (unsigned short*)smem;                  // [512][256] bf16
    float* hbuf = (float*)(smem + (size_t)WLSTM_LDS_ROWS * HP * 2);
    float* cbuf = hbuf + HP;
    float* gbuf = cbuf + HP;                                       // [1024]
    const int tid = threadIdx.x;

    {   // stage first 512 Whh rows into LDS (one time)
        v4u*       dst = (v4u*)wlds;
        const v4u* src = (const v4u*)whh;
        for (int i = tid; i < WLSTM_LDS_ROWS * HP / 8; i += 1024) dst[i] = src[i];
    }
    if (tid < HP) { hbuf[tid] = 0.f; cbuf[tid] = 0.f; }
    __syncthreads();

    const v4u* wl = (const v4u*)wlds + (size_t)tid * (HP / 8);
    const v4u* wg = (const v4u*)whh  + (size_t)tid * (HP / 8);

    for (int t = 0; t < NW; ++t) {
        float x = xproj[(size_t)t * 1024 + tid];
        v2f a2 = {0.f, 0.f};
        if (tid < WLSTM_LDS_ROWS) {
#pragma unroll 8
            for (int j = 0; j < HP / 8; ++j) {
                v4u u  = wl[j];
                v4f h0 = *reinterpret_cast<const v4f*>(hbuf + j * 8);
                v4f h1 = *reinterpret_cast<const v4f*>(hbuf + j * 8 + 4);
                a2 = pkfma(u.x, h0.x, h0.y, a2);
                a2 = pkfma(u.y, h0.z, h0.w, a2);
                a2 = pkfma(u.z, h1.x, h1.y, a2);
                a2 = pkfma(u.w, h1.z, h1.w, a2);
            }
        } else {
#pragma unroll 8
            for (int j = 0; j < HP / 8; ++j) {
                v4u u  = wg[j];
                v4f h0 = *reinterpret_cast<const v4f*>(hbuf + j * 8);
                v4f h1 = *reinterpret_cast<const v4f*>(hbuf + j * 8 + 4);
                a2 = pkfma(u.x, h0.x, h0.y, a2);
                a2 = pkfma(u.y, h0.z, h0.w, a2);
                a2 = pkfma(u.z, h1.x, h1.y, a2);
                a2 = pkfma(u.w, h1.z, h1.w, a2);
            }
        }
        gbuf[tid] = x + a2.x + a2.y;
        __syncthreads();
        if (tid < HP) {
            float iv = sigm_fast(gbuf[tid]);
            float fv = sigm_fast(gbuf[HP + tid]);
            float gv = tanh_fast(gbuf[2 * HP + tid]);
            float ov = sigm_fast(gbuf[3 * HP + tid]);
            float cn = fv * cbuf[tid] + iv * gv;
            float hn = ov * tanh_fast(cn);
            cbuf[tid] = cn;
            hbuf[tid] = hn;
            hs[(size_t)t * HP + tid] = hn;
        }
        __syncthreads();
    }
}

// ---------------------------------------------------------------------------
// Phase D: tag projection + log_softmax.  4 words per block, 64 lanes/word.
// ---------------------------------------------------------------------------
__global__ __launch_bounds__(256) void tag_kernel(
    const float* __restrict__ hs, const float* __restrict__ W_tag,
    const float* __restrict__ b_tag, float* __restrict__ out) {

    __shared__ float red[4][64];
    const int tid = threadIdx.x;
    const int grp = tid >> 6;
    const int l   = tid & 63;
    const int w   = blockIdx.x * 4 + grp;

    float val = 0.f;
    if (l < TAGS) {
        val = b_tag[l];
        const float* hr = hs + (size_t)w * HP;
        const float* wt = W_tag + (size_t)l * HP;
#pragma unroll 4
        for (int k = 0; k < HP; ++k) val += hr[k] * wt[k];
    }
    red[grp][l] = (l < TAGS) ? val : -1e30f;
    __syncthreads();
#pragma unroll
    for (int s = 32; s > 0; s >>= 1) {
        if (l < s) red[grp][l] = fmaxf(red[grp][l], red[grp][l + s]);
        __syncthreads();
    }
    float mx = red[grp][0];
    __syncthreads();
    red[grp][l] = (l < TAGS) ? __expf(val - mx) : 0.f;
    __syncthreads();
#pragma unroll
    for (int s = 32; s > 0; s >>= 1) {
        if (l < s) red[grp][l] += red[grp][l + s];
        __syncthreads();
    }
    float sm = red[grp][0];
    if (l < TAGS) out[(size_t)w * TAGS + l] = val - mx - __logf(sm);
}

// ---------------------------------------------------------------------------
// Host launcher
// ---------------------------------------------------------------------------
extern "C" void kernel_launch(void* const* d_in, const int* in_sizes, int n_in,
                              void* d_out, int out_size, void* d_ws, size_t ws_size,
                              hipStream_t stream) {
    const float* char_emb = (const float*)d_in[0];
    const float* word_emb = (const float*)d_in[1];
    const float* Wih_e    = (const float*)d_in[2];
    const float* Whh_e    = (const float*)d_in[3];
    const float* bih_e    = (const float*)d_in[4];
    const float* bhh_e    = (const float*)d_in[5];
    const float* Wih_p    = (const float*)d_in[6];
    const float* Whh_p    = (const float*)d_in[7];
    const float* bih_p    = (const float*)d_in[8];
    const float* bhh_p    = (const float*)d_in[9];
    const float* W_tag    = (const float*)d_in[10];
    const float* b_tag    = (const float*)d_in[11];
    const int*   sentence = (const int*)d_in[12];
    const int*   char_ids = (const int*)d_in[13];
    const int*   char_len = (const int*)d_in[14];
    float*       out      = (float*)d_out;

    char* ws = (char*)d_ws;
    float*          h_char  = (float*)(ws + OFF_HCHAR);
    unsigned short* wfrag_e = (unsigned short*)(ws + OFF_WFE);
    float*          bias_e  = (float*)(ws + OFF_BE);
    unsigned short* feats   = (unsigned short*)(ws + OFF_FEATS);
    unsigned short* wfrag_p = (unsigned short*)(ws + OFF_WFP);
    float*          bias_p  = (float*)(ws + OFF_BP);
    unsigned short* whh_bf  = (unsigned short*)(ws + OFF_WHH);
    float*          xproj   = (float*)(ws + OFF_XPROJ);
    float*          hs      = (float*)(ws + OFF_HS);

    prep_wfrag_e<<<384, 256, 0, stream>>>(Wih_e, Whh_e, wfrag_e);
    prep_wfrag_p<<<1536, 256, 0, stream>>>(Wih_p, wfrag_p);
    prep_misc<<<1024, 256, 0, stream>>>(Whh_p, bih_e, bhh_e, bih_p, bhh_p,
                                        whh_bf, bias_e, bias_p);

    char_lstm_kernel<<<NW / 16, 256, 0, stream>>>(char_emb, char_ids, char_len,
                                                  wfrag_e, bias_e, h_char);

    feats_kernel<<<(NW * KP / 4) / 256, 256, 0, stream>>>(word_emb, h_char,
                                                          sentence, feats);

    xproj_kernel<<<(NW / 16) * 8, 256, 0, stream>>>(feats, wfrag_p, bias_p, xproj);

    word_lstm_kernel<<<1, 1024, WLSTM_SMEM, stream>>>(xproj, whh_bf, hs);

    tag_kernel<<<NW / 4, 256, 0, stream>>>(hs, W_tag, b_tag, out);
}
